// TopKSAE_52055003628278
// MI455X (gfx1250) — matrix operand order repaired and checked by reading, not compiled, not run on hardware
//
#include <hip/hip_runtime.h>
#include <stdint.h>

#define INP_DIM 1024
#define HID_DIM 16384
#define NTOK    8192
#define TOPK    64

typedef __attribute__((ext_vector_type(16))) __bf16 v16bf;
typedef __attribute__((ext_vector_type(8)))  float  v8f;

// ---------------- helpers ----------------
__device__ __forceinline__ unsigned short f2bf_rne(float x) {
    unsigned u = __float_as_uint(x);
    unsigned r = u + 0x7FFFu + ((u >> 16) & 1u);
    return (unsigned short)(r >> 16);
}
__device__ __forceinline__ float bf2f(unsigned short h) {
    return __uint_as_float(((unsigned)h) << 16);
}

union Frag {
    v16bf v;
    uint4 q[2];
};

// ---------------- kernel 1: W_dec transpose (1024 x 16384 -> 16384 x 1024) ----------------
__global__ __launch_bounds__(256) void sae_transpose_wdec(const float* __restrict__ W,
                                                          float* __restrict__ WT) {
    __shared__ float t[32][33];
    const int bx = blockIdx.x;          // along HID (512 tiles)
    const int by = blockIdx.y;          // along INP (32 tiles)
    const int tx = threadIdx.x & 31;
    const int ty = threadIdx.x >> 5;    // 0..7
#pragma unroll
    for (int j = 0; j < 4; ++j) {
        t[ty + 8 * j][tx] = W[(size_t)(by * 32 + ty + 8 * j) * HID_DIM + bx * 32 + tx];
    }
    __syncthreads();
#pragma unroll
    for (int j = 0; j < 4; ++j) {
        WT[(size_t)(bx * 32 + ty + 8 * j) * INP_DIM + by * 32 + tx] = t[tx][ty + 8 * j];
    }
}

// ---------------- kernel 2: encoder GEMM via bf16x2-split WMMA, double-buffered LDS ----------------
// enc[M=8192, N=16384] = emb[M,K=1024] @ W_enc[N,K]^T + b_enc
// Block tile 128x128, 8 waves (2x4), each wave 64x32 = 4x2 WMMA tiles, K-step 32, ping-pong LDS.
#define LDS_ROW_B 80                         // 32 bf16 = 64B payload + 16B pad
#define LDS_BANK  (4 * 128 * LDS_ROW_B)      // Ahi|Alo|Bhi|Blo regions = 40960 B per buffer

struct Stage {
    float a[16];
    float b[16];
};

__device__ __forceinline__ void load_slab(const float* __restrict__ aSrc,
                                          const float* __restrict__ bSrc,
                                          int kk, Stage& s) {
#pragma unroll
    for (int j = 0; j < 4; ++j) {
        float4 a4 = *(const float4*)(aSrc + kk + j * 4);
        float4 b4 = *(const float4*)(bSrc + kk + j * 4);
        s.a[4 * j + 0] = a4.x; s.a[4 * j + 1] = a4.y; s.a[4 * j + 2] = a4.z; s.a[4 * j + 3] = a4.w;
        s.b[4 * j + 0] = b4.x; s.b[4 * j + 1] = b4.y; s.b[4 * j + 2] = b4.z; s.b[4 * j + 3] = b4.w;
    }
}

__device__ __forceinline__ void convert_store(uint32_t* __restrict__ aHiRow,
                                              uint32_t* __restrict__ aLoRow,
                                              uint32_t* __restrict__ bHiRow,
                                              uint32_t* __restrict__ bLoRow,
                                              const Stage& s) {
#pragma unroll
    for (int j = 0; j < 8; ++j) {
        unsigned short ah0 = f2bf_rne(s.a[2 * j]),     ah1 = f2bf_rne(s.a[2 * j + 1]);
        unsigned short bh0 = f2bf_rne(s.b[2 * j]),     bh1 = f2bf_rne(s.b[2 * j + 1]);
        unsigned short al0 = f2bf_rne(s.a[2 * j]     - bf2f(ah0));
        unsigned short al1 = f2bf_rne(s.a[2 * j + 1] - bf2f(ah1));
        unsigned short bl0 = f2bf_rne(s.b[2 * j]    - bf2f(bh0));
        unsigned short bl1 = f2bf_rne(s.b[2 * j + 1] - bf2f(bh1));
        aHiRow[j] = (unsigned)ah0 | ((unsigned)ah1 << 16);
        aLoRow[j] = (unsigned)al0 | ((unsigned)al1 << 16);
        bHiRow[j] = (unsigned)bh0 | ((unsigned)bh1 << 16);
        bLoRow[j] = (unsigned)bl0 | ((unsigned)bl1 << 16);
    }
}

__global__ __launch_bounds__(256) void sae_encode_wmma(const float* __restrict__ emb,
                                                       const float* __restrict__ Wenc,
                                                       const float* __restrict__ benc,
                                                       float* __restrict__ enc) {
    __shared__ __align__(16) unsigned char lds[2 * LDS_BANK]; // 80 KB ping-pong

    const int tid  = threadIdx.x;
    const int lane = tid & 31;
    const int wave = tid >> 5;
    const int wm   = wave & 1;   // 0..1 : 64-row band
    const int wn   = wave >> 1;  // 0..3 : 32-col band
    const int r    = lane & 15;
    const int half = lane >> 4;

    const int n0 = blockIdx.x * 128;
    const int m0 = blockIdx.y * 128;

    // loader mapping: 256 threads cover 128 rows x 32 K-floats (16 floats per thread)
    const int lr = tid >> 1;   // 0..127
    const int hk = tid & 1;    // which 16-float half of the K-slab
    const float* aSrc = emb  + (size_t)(m0 + lr) * INP_DIM + hk * 16;
    const float* bSrc = Wenc + (size_t)(n0 + lr) * INP_DIM + hk * 16;

    // per-buffer LDS row pointers for the loader
    uint32_t* aHiRowB[2];
    uint32_t* aLoRowB[2];
    uint32_t* bHiRowB[2];
    uint32_t* bLoRowB[2];
#pragma unroll
    for (int b = 0; b < 2; ++b) {
        unsigned char* bank = lds + b * LDS_BANK;
        aHiRowB[b] = (uint32_t*)(bank + 0 * 128 * LDS_ROW_B + lr * LDS_ROW_B) + hk * 8;
        aLoRowB[b] = (uint32_t*)(bank + 1 * 128 * LDS_ROW_B + lr * LDS_ROW_B) + hk * 8;
        bHiRowB[b] = (uint32_t*)(bank + 2 * 128 * LDS_ROW_B + lr * LDS_ROW_B) + hk * 8;
        bLoRowB[b] = (uint32_t*)(bank + 3 * 128 * LDS_ROW_B + lr * LDS_ROW_B) + hk * 8;
    }

    const v8f vzero = {0.f, 0.f, 0.f, 0.f, 0.f, 0.f, 0.f, 0.f};
    v8f acc[4][2];
#pragma unroll
    for (int mt = 0; mt < 4; ++mt)
#pragma unroll
        for (int nt = 0; nt < 2; ++nt) acc[mt][nt] = vzero;

    // ---- prologue: stage K-slab 0 into buffer 0 ----
    Stage st;
    load_slab(aSrc, bSrc, 0, st);
    convert_store(aHiRowB[0], aLoRowB[0], bHiRowB[0], bLoRowB[0], st);
    __syncthreads();

    const int NITER = INP_DIM / 32;
    for (int it = 0; it < NITER; ++it) {
        const int buf = it & 1;
        const unsigned char* bank = lds + buf * LDS_BANK;
        const unsigned char* Ahi = bank;
        const unsigned char* Alo = bank + 1 * 128 * LDS_ROW_B;
        const unsigned char* Bhi = bank + 2 * 128 * LDS_ROW_B;
        const unsigned char* Blo = bank + 3 * 128 * LDS_ROW_B;

        // issue next slab's global loads early so vmem latency overlaps WMMA
        if (it + 1 < NITER) {
            load_slab(aSrc, bSrc, (it + 1) * 32, st);
            if (it + 2 < NITER) {
                __builtin_prefetch(aSrc + (it + 2) * 32, 0, 1);
                __builtin_prefetch(bSrc + (it + 2) * 32, 0, 1);
            }
        }

        // ---- build fragments from LDS ----
        // A layout (ISA 7.12.2, 16-bit A 16x32): lanes 0-15 hold K{0..7,16..23}, lanes 16-31 K{8..15,24..31}
        Frag aH[4], aL[4];
#pragma unroll
        for (int mt = 0; mt < 4; ++mt) {
            const unsigned char* rh = Ahi + (wm * 64 + mt * 16 + r) * LDS_ROW_B;
            const unsigned char* rl = Alo + (wm * 64 + mt * 16 + r) * LDS_ROW_B;
            aH[mt].q[0] = *(const uint4*)(rh + half * 16);
            aH[mt].q[1] = *(const uint4*)(rh + 32 + half * 16);
            aL[mt].q[0] = *(const uint4*)(rl + half * 16);
            aL[mt].q[1] = *(const uint4*)(rl + 32 + half * 16);
        }
        // B layout (same K order as A): lanes 0-15 K{0..7,16..23}, lanes 16-31 K{8..15,24..31}
        Frag bH[2], bL[2];
#pragma unroll
        for (int nt = 0; nt < 2; ++nt) {
            const unsigned char* rh = Bhi + (wn * 32 + nt * 16 + r) * LDS_ROW_B;
            const unsigned char* rl = Blo + (wn * 32 + nt * 16 + r) * LDS_ROW_B;
            bH[nt].q[0] = *(const uint4*)(rh + half * 16);
            bH[nt].q[1] = *(const uint4*)(rh + 32 + half * 16);
            bL[nt].q[0] = *(const uint4*)(rl + half * 16);
            bL[nt].q[1] = *(const uint4*)(rl + 32 + half * 16);
        }

        // ---- 3-term split-bf16 WMMA:  AhBh + AhBl + AlBh ----
#pragma unroll
        for (int mt = 0; mt < 4; ++mt) {
#pragma unroll
            for (int nt = 0; nt < 2; ++nt) {
                acc[mt][nt] = __builtin_amdgcn_wmma_f32_16x16x32_bf16(
                    false, aH[mt].v, false, bH[nt].v, (short)0, acc[mt][nt], false, false);
                acc[mt][nt] = __builtin_amdgcn_wmma_f32_16x16x32_bf16(
                    false, aH[mt].v, false, bL[nt].v, (short)0, acc[mt][nt], false, false);
                acc[mt][nt] = __builtin_amdgcn_wmma_f32_16x16x32_bf16(
                    false, aL[mt].v, false, bH[nt].v, (short)0, acc[mt][nt], false, false);
            }
        }

        // ---- convert + store next slab into the other buffer (VALU co-executes with WMMA) ----
        if (it + 1 < NITER) {
            const int nb = buf ^ 1;
            convert_store(aHiRowB[nb], aLoRowB[nb], bHiRowB[nb], bLoRowB[nb], st);
        }
        __syncthreads();
    }

    // ---- epilogue: add bias, store C (lane: N=l&15; VGPR v: M=v+8*(l>>4)) ----
#pragma unroll
    for (int nt = 0; nt < 2; ++nt) {
        const int n = n0 + wn * 32 + nt * 16 + r;
        const float bias = benc[n];
#pragma unroll
        for (int mt = 0; mt < 4; ++mt) {
            const int mbase = m0 + wm * 64 + mt * 16 + 8 * half;
#pragma unroll
            for (int v = 0; v < 8; ++v) {
                enc[(size_t)(mbase + v) * HID_DIM + n] = acc[mt][nt][v] + bias;
            }
        }
    }
}

// ---------------- kernel 3: per-row top-64 by |value| ----------------
__global__ __launch_bounds__(256) void sae_topk(const float* __restrict__ enc,
                                                int* __restrict__ idxL,
                                                float* __restrict__ valL) {
    const int row = blockIdx.x;
    const int tid = threadIdx.x;
    const float* rp = enc + (size_t)row * HID_DIM;

    float v[64];
#pragma unroll
    for (int i = 0; i < 64; ++i) v[i] = rp[i * 256 + tid];  // gidx = i*256 + tid

    unsigned long long dead = 0ull;
    __shared__ float wA[8];
    __shared__ float wV[8];
    __shared__ int   wG[8];
    __shared__ int   winG;

    for (int round = 0; round < TOPK; ++round) {
        float bA = -1.0f, bV = 0.0f;
        int   bG = 0x7fffffff;
#pragma unroll
        for (int i = 0; i < 64; ++i) {
            if (!((dead >> i) & 1ull)) {
                float a = fabsf(v[i]);
                int   g = i * 256 + tid;
                if (a > bA || (a == bA && g < bG)) { bA = a; bV = v[i]; bG = g; }
            }
        }
        // wave32 reduction
#pragma unroll
        for (int off = 16; off > 0; off >>= 1) {
            float oa = __shfl_xor(bA, off, 32);
            float ov = __shfl_xor(bV, off, 32);
            int   og = __shfl_xor(bG, off, 32);
            if (oa > bA || (oa == bA && og < bG)) { bA = oa; bV = ov; bG = og; }
        }
        if ((tid & 31) == 0) { wA[tid >> 5] = bA; wV[tid >> 5] = bV; wG[tid >> 5] = bG; }
        __syncthreads();
        if (tid == 0) {
            float fA = wA[0]; float fV = wV[0]; int fG = wG[0];
#pragma unroll
            for (int w = 1; w < 8; ++w)
                if (wA[w] > fA || (wA[w] == fA && wG[w] < fG)) { fA = wA[w]; fV = wV[w]; fG = wG[w]; }
            idxL[row * TOPK + round] = fG;
            valL[row * TOPK + round] = fV;
            winG = fG;
        }
        __syncthreads();
        const int g = winG;
        if ((g & 255) == tid) dead |= (1ull << (g >> 8));
    }
}

// ---------------- kernel 4: sparse decode ----------------
// recon[row, :] = b_dec + sum_j val[j] * Wdec[:, idx[j]]
// use_wt: WT is W_dec^T [HID, INP] (coalesced rows); else read W_dec [INP, HID] strided.
__global__ __launch_bounds__(256) void sae_decode(const int* __restrict__ idxL,
                                                  const float* __restrict__ valL,
                                                  const float* __restrict__ WT,
                                                  const float* __restrict__ Wdec,
                                                  const float* __restrict__ bdec,
                                                  float* __restrict__ recon,
                                                  int use_wt) {
    const int row = blockIdx.x;
    const int tid = threadIdx.x;
    __shared__ int   sIdx[TOPK];
    __shared__ float sVal[TOPK];
    if (tid < TOPK) {
        sIdx[tid] = idxL[row * TOPK + tid];
        sVal[tid] = valL[row * TOPK + tid];
    }
    __syncthreads();

    float acc[4];
#pragma unroll
    for (int p = 0; p < 4; ++p) acc[p] = bdec[p * 256 + tid];

    if (use_wt) {
        for (int j = 0; j < TOPK; ++j) {
            const float* wrow = WT + (size_t)sIdx[j] * INP_DIM;
            const float  s    = sVal[j];
#pragma unroll
            for (int p = 0; p < 4; ++p) acc[p] += s * wrow[p * 256 + tid];
        }
    } else {
        for (int j = 0; j < TOPK; ++j) {
            const int   c = sIdx[j];
            const float s = sVal[j];
#pragma unroll
            for (int p = 0; p < 4; ++p)
                acc[p] += s * Wdec[(size_t)(p * 256 + tid) * HID_DIM + c];
        }
    }
    float* out = recon + (size_t)row * INP_DIM;
#pragma unroll
    for (int p = 0; p < 4; ++p) out[p * 256 + tid] = acc[p];
}

// ---------------- launch ----------------
extern "C" void kernel_launch(void* const* d_in, const int* in_sizes, int n_in,
                              void* d_out, int out_size, void* d_ws, size_t ws_size,
                              hipStream_t stream) {
    const float* emb  = (const float*)d_in[0];
    const float* Wenc = (const float*)d_in[1];
    const float* benc = (const float*)d_in[2];
    const float* Wdec = (const float*)d_in[3];
    const float* bdec = (const float*)d_in[4];

    float* enc   = (float*)d_out;                         // [8192, 16384]
    float* recon = enc + (size_t)NTOK * HID_DIM;          // [8192, 1024]

    char*  ws   = (char*)d_ws;
    int*   idxL = (int*)ws;                                           // 2 MB
    float* valL = (float*)(ws + (size_t)NTOK * TOPK * sizeof(int));   // 2 MB
    const size_t listBytes = (size_t)NTOK * TOPK * 8;
    const size_t wtBytes   = (size_t)HID_DIM * INP_DIM * sizeof(float); // 64 MB
    const int use_wt = (ws_size >= listBytes + wtBytes) ? 1 : 0;
    float* WT = (float*)(ws + listBytes);

    if (use_wt) {
        sae_transpose_wdec<<<dim3(HID_DIM / 32, INP_DIM / 32), 256, 0, stream>>>(Wdec, WT);
    }
    sae_encode_wmma<<<dim3(HID_DIM / 128, NTOK / 128), 256, 0, stream>>>(emb, Wenc, benc, enc);
    sae_topk<<<NTOK, 256, 0, stream>>>(enc, idxL, valL);
    sae_decode<<<NTOK, 256, 0, stream>>>(idxL, valL, WT, Wdec, bdec, recon, use_wt);
}
